// mmse_mamba_12799002542236
// MI455X (gfx1250) — compile-verified, matrix operand order checked by reading, not run on hardware
//
#include <hip/hip_runtime.h>
#include <math.h>

typedef __bf16 bf16;
typedef __attribute__((ext_vector_type(16))) __bf16 v16bf;
typedef __attribute__((ext_vector_type(8)))  __bf16 v8bf;
typedef __attribute__((ext_vector_type(8)))  float  v8f;
typedef __attribute__((ext_vector_type(4)))  float  v4f;

#define BB 8
#define CC 256
#define HH 64
#define WW 64
#define H2 128
#define W2 128
#define LL 128
#define DI 512
#define NST 16

static __device__ __forceinline__ float sigf(float x){ return 1.f/(1.f+expf(-x)); }

static __device__ __forceinline__ v8bf zero8(){
  v8bf z;
  #pragma unroll
  for (int i=0;i<8;i++) z[i] = (bf16)0.f;
  return z;
}

// load 8 contiguous f32 (16B-aligned) and convert to v8bf
static __device__ __forceinline__ v8bf ld8f32_bf16(const float* __restrict__ p){
  const v4f* q = (const v4f*)p;
  v4f a = q[0], b = q[1];
  v8bf r;
  r[0]=(bf16)a[0]; r[1]=(bf16)a[1]; r[2]=(bf16)a[2]; r[3]=(bf16)a[3];
  r[4]=(bf16)b[0]; r[5]=(bf16)b[1]; r[6]=(bf16)b[2]; r[7]=(bf16)b[3];
  return r;
}

#define CAT16(lo,hi) __builtin_shufflevector(lo,hi,0,1,2,3,4,5,6,7,8,9,10,11,12,13,14,15)

// ---------------------------------------------------------------------------
// weight f32 -> bf16 conversion
__global__ void k_cvt_bf16(const float* __restrict__ s, bf16* __restrict__ d, int n){
  int i = blockIdx.x*256 + threadIdx.x;
  if (i < n) d[i] = (bf16)s[i];
}

// ---------------------------------------------------------------------------
// depthwise transposed conv 3x3 stride 2 : x(8,256,64,64) -> z0(8,256,128,128)
__global__ void k_dwconvT(const float* __restrict__ x, const float* __restrict__ w,
                          float* __restrict__ z0){
  int idx = blockIdx.x*256 + threadIdx.x;                 // 33,554,432 total
  int xj = idx & 127, yi = (idx>>7)&127, c = (idx>>14)&255, b = idx>>22;
  if (b >= BB) return;
  const float* xp = x + (((size_t)b*CC + c)*HH)*WW;
  const float* wc = w + c*9;
  float acc = 0.f;
  #pragma unroll
  for (int kh=0; kh<3; ++kh){
    int ty = yi + kh - 1;
    if (ty & 1) continue;                                  // lhs_dilation=2 parity
    int sy = ty >> 1;
    if (ty < 0 || sy >= HH) continue;
    #pragma unroll
    for (int kw=0; kw<3; ++kw){
      int tx = xj + kw - 1;
      if (tx & 1) continue;
      int sx = tx >> 1;
      if (tx < 0 || sx >= WW) continue;
      acc += wc[(2-kh)*3 + (2-kw)] * xp[sy*WW + sx];       // kernel pre-flipped in ref
    }
  }
  z0[idx] = acc;
}

// ---------------------------------------------------------------------------
// per-channel BN stats over (B,pix): writes a=g*rstd, bsh=beta-mean*a
__global__ void k_chan_stats(const float* __restrict__ d, int pix,
                             const float* __restrict__ g, const float* __restrict__ be,
                             float* __restrict__ a, float* __restrict__ bs){
  int c = blockIdx.x, t = threadIdx.x;
  __shared__ float s1[256], s2[256];
  float su = 0.f, sq = 0.f;
  for (int b=0;b<BB;b++){
    const float* p = d + ((size_t)b*CC + c)*pix;
    for (int i=t;i<pix;i+=256){ float v=p[i]; su+=v; sq+=v*v; }
  }
  s1[t]=su; s2[t]=sq; __syncthreads();
  for (int o=128;o>0;o>>=1){ if (t<o){ s1[t]+=s1[t+o]; s2[t]+=s2[t+o]; } __syncthreads(); }
  if (t==0){
    float n  = (float)pix * (float)BB;
    float m  = s1[0]/n;
    float vr = s2[0]/n - m*m;
    float rs = rsqrtf(vr + 1e-5f);
    float av = g[c]*rs;
    a[c]=av; bs[c]=be[c]-m*av;
  }
}

// per-channel affine in place; MODE 0: plain, 2: clip[0,6]
template<int MODE>
__global__ void k_affine(float* __restrict__ d, const float* __restrict__ a,
                         const float* __restrict__ bs, int pix){
  size_t i = (size_t)blockIdx.x*256 + threadIdx.x;
  size_t total = (size_t)BB*CC*pix;
  if (i >= total) return;
  int c = (int)((i/(size_t)pix) % CC);
  float v = a[c]*d[i] + bs[c];
  if (MODE==2){ v = v<0.f?0.f:(v>6.f?6.f:v); }
  d[i] = v;
}

// ---------------------------------------------------------------------------
// NCHW pointwise-conv GEMM with WMMA: out[b,o,p] = sum_c W[o,c]*T(act[b,c,p])
// MODE 1: T = relu(a*v+b) (fused BN1+ReLU), MODE 0: identity (compile-time).
// Block: 256 thr = 8 waves (2 M x 4 N); block tile M=32, N=256 pixels; K=CC.
// LDS tile stored transposed [N=256][K=32 pad->40] so B fragments are two
// contiguous 16B ds_load_b128 per lane; fill is 4x ds_store_b128 per thread.
template<int MODE>
__global__ void k_gemm_nchw(const float* __restrict__ act, const bf16* __restrict__ Wb,
                            float* __restrict__ out, const float* __restrict__ ta,
                            const float* __restrict__ tb, int pix){
  __shared__ bf16 lb[256][40];                             // 80B rows, 16B aligned
  int tix = threadIdx.x;
  int wv = tix>>5, lane = tix&31, hi = lane>>4, lr = lane&15;
  int wm = wv>>2, wn = wv&3;
  int m0 = blockIdx.x*32 + wm*16;
  int q0 = blockIdx.y*256;
  int b  = q0/pix, p0 = q0 - b*pix;
  const float* actb = act + (size_t)b*CC*pix;
  v8f accs[4];
  #pragma unroll
  for (int t=0;t<4;t++){
    #pragma unroll
    for (int i=0;i<8;i++) accs[t][i]=0.f;
  }
  for (int k0=0;k0<CC;k0+=32){
    __syncthreads();
    {
      // each thread owns one pixel column: coalesced global reads across lanes
      const float* srcc = actb + (size_t)k0*pix + p0 + tix;
      if (k0+32 < CC)
        __builtin_prefetch(actb + (size_t)(k0+32)*pix + p0 + tix, 0, 1);
      v8bf r0=zero8(), r1=zero8(), r2=zero8(), r3=zero8();
      #pragma unroll
      for (int k=0;k<32;k++){
        float v = srcc[(size_t)k*pix];
        if (MODE==1){
          v = ta[k0+k]*v + tb[k0+k];
          v = v>0.f?v:0.f;
        }
        if      (k<8)  r0[k]    = (bf16)v;
        else if (k<16) r1[k-8]  = (bf16)v;
        else if (k<24) r2[k-16] = (bf16)v;
        else           r3[k-24] = (bf16)v;
      }
      v8bf* lp = (v8bf*)&lb[tix][0];
      lp[0]=r0; lp[1]=r1; lp[2]=r2; lp[3]=r3;
    }
    __syncthreads();
    // A fragment: weights, two contiguous 16B runs per lane (ISA 16-bit layout)
    const bf16* wrow = Wb + (size_t)(m0+lr)*CC + k0;
    v8bf alo = *(const v8bf*)(wrow + hi*8);
    v8bf ahi = *(const v8bf*)(wrow + 16 + hi*8);
    v16bf av = CAT16(alo, ahi);
    #pragma unroll
    for (int t=0;t<4;t++){
      int n = wn*64 + t*16 + lr;
      const v8bf* lrp = (const v8bf*)&lb[n][0];
      v8bf blo = lrp[hi];                                  // K rel hi*8..+7
      v8bf bhi = lrp[2+hi];                                // K rel 16+hi*8..+7
      v16bf bv = CAT16(blo, bhi);
      accs[t] = __builtin_amdgcn_wmma_f32_16x16x32_bf16(
                  false, av, false, bv, (short)0, accs[t], false, false);
    }
  }
  float* outb = out + (size_t)b*CC*pix;
  #pragma unroll
  for (int t=0;t<4;t++){
    int p = p0 + wn*64 + t*16 + lr;                        // lane holds fixed N col
    #pragma unroll
    for (int rr=0;rr<8;rr++){
      int m = m0 + hi*8 + rr;                              // C/D f32 layout
      outb[(size_t)m*pix + p] = accs[t][rr];
    }
  }
}

// ---------------------------------------------------------------------------
// Sequence GEMM (K contiguous both operands): out[n,m] = sum_k act[n*ldA+k]*W[m*K+k]
// EPI 0: none, 1: softplus(v+bias[m]), 2: v+bias[m]   (compile-time)
// All fragment loads vectorized: v8bf weight loads, 2x float4 activation loads.
template<int EPI>
__global__ void k_gemm_seq(const float* __restrict__ act, const bf16* __restrict__ Wb,
                           float* __restrict__ out, int Ntot, int Mtot, int K, int ldA,
                           const float* __restrict__ bias){
  int ntile = Ntot>>4, mtile = Mtot>>4;
  int tid = blockIdx.x*8 + (threadIdx.x>>5);
  if (tid >= ntile*mtile) return;                          // wave-uniform exit
  int tn = tid % ntile, tm = tid / ntile;
  int lane = threadIdx.x&31, hi = lane>>4, lr = lane&15;
  int n0 = tn*16, m0 = tm*16;
  v8f acc;
  #pragma unroll
  for (int i=0;i<8;i++) acc[i]=0.f;
  const bf16*  wrow = Wb  + (size_t)(m0+lr)*K;
  const float* arow = act + (size_t)(n0+lr)*ldA;
  v8bf z8 = zero8();
  for (int k0=0;k0<K;k0+=32){
    int kb0 = k0 + hi*8;                                   // 8-run bases; runs never
    int kb1 = k0 + 16 + hi*8;                              // straddle K (K % 16 == 0)
    v8bf alo = (kb0 < K) ? *(const v8bf*)(wrow + kb0) : z8;
    v8bf ahi = (kb1 < K) ? *(const v8bf*)(wrow + kb1) : z8;
    v8bf blo = (kb0 < K) ? ld8f32_bf16(arow + kb0) : z8;
    v8bf bhi = (kb1 < K) ? ld8f32_bf16(arow + kb1) : z8;
    v16bf av = CAT16(alo, ahi);
    v16bf bv = CAT16(blo, bhi);
    acc = __builtin_amdgcn_wmma_f32_16x16x32_bf16(
            false, av, false, bv, (short)0, acc, false, false);
  }
  int n = n0 + lr;
  float* op = out + (size_t)n*Mtot + m0 + hi*8;            // contiguous 8-float run
  v4f o0, o1;
  #pragma unroll
  for (int rr=0;rr<8;rr++){
    float v = acc[rr];
    int m = m0 + hi*8 + rr;
    if (EPI==1){ v += bias[m]; v = (v>20.f)? v : log1pf(expf(v)); }
    else if (EPI==2){ v += bias[m]; }
    if (rr<4) o0[rr] = v; else o1[rr-4] = v;
  }
  *(v4f*)op       = o0;
  *(v4f*)(op + 4) = o1;
}

// ---------------------------------------------------------------------------
// row/col max+min reductions + argmax/argmin + resized pos-embed + LayerNorm
// one block per (b,l); l<64 -> column reduce (over h), l>=64 -> row reduce (over w)
__global__ void k_reduce_axial(const float* __restrict__ x, const float* __restrict__ pos,
                               const float* __restrict__ lw, const float* __restrict__ lb2,
                               float* __restrict__ rmax, float* __restrict__ rmin){
  int bid = blockIdx.x, b = bid>>7, l = bid&127, c = threadIdx.x;
  const float* xb = x + ((size_t)b*CC + c)*HH*WW;
  float mx=-3.4e38f, mn=3.4e38f, ima=0.f, imi=0.f;
  int j;
  if (l < 64){
    j = l;
    for (int h=0;h<HH;h++){
      float v = xb[h*WW + l];
      if (v>mx){mx=v; ima=(float)h;}
      if (v<mn){mn=v; imi=(float)h;}
    }
  } else {
    j = l-64;
    const float* rp = xb + (size_t)(l-64)*WW;
    for (int w=0;w<WW;w++){
      float v = rp[w];
      if (v>mx){mx=v; ima=(float)w;}
      if (v<mn){mn=v; imi=(float)w;}
    }
  }
  // linear resize pos(1,C,16) -> 64, half-pixel centers, edge clamp
  float t  = (j+0.5f)*0.25f - 0.5f;
  float fl = floorf(t), fr = t-fl;
  int i0 = (int)fl, i1 = i0+1;
  i0 = i0<0?0:(i0>15?15:i0);
  i1 = i1<0?0:(i1>15?15:i1);
  float pe = pos[c*16+i0]*(1.f-fr) + pos[c*16+i1]*fr;
  float vmax = mx + pe + ima;
  float vmin = mn + pe + imi;
  __shared__ float s1[256], s2[256];
  // LN over channels (max)
  s1[c]=vmax; s2[c]=vmax*vmax; __syncthreads();
  for (int o=128;o>0;o>>=1){ if (c<o){s1[c]+=s1[c+o]; s2[c]+=s2[c+o];} __syncthreads(); }
  float m  = s1[0]*(1.f/256.f);
  float vr = s2[0]*(1.f/256.f)-m*m;
  float rs = rsqrtf(vr+1e-5f);
  rmax[((size_t)b*LL+l)*CC+c] = (vmax-m)*rs*lw[c]+lb2[c];
  __syncthreads();
  // LN (min)
  s1[c]=vmin; s2[c]=vmin*vmin; __syncthreads();
  for (int o=128;o>0;o>>=1){ if (c<o){s1[c]+=s1[c+o]; s2[c]+=s2[c+o];} __syncthreads(); }
  m  = s1[0]*(1.f/256.f);
  vr = s2[0]*(1.f/256.f)-m*m;
  rs = rsqrtf(vr+1e-5f);
  rmin[((size_t)b*LL+l)*CC+c] = (vmin-m)*rs*lw[c]+lb2[c];
}

// flip along L: (B,L,C)
__global__ void k_flip_l(const float* __restrict__ s, float* __restrict__ d){
  int i = blockIdx.x*256 + threadIdx.x;
  if (i >= BB*LL*CC) return;
  int c = i & 255, l = (i>>8)&127, b = i>>15;
  d[i] = s[((size_t)b*LL + (LL-1-l))*CC + c];
}

// causal depthwise conv1d (k=4) + bias + silu over xm = xz[...,:512]
__global__ void k_conv1d_silu(const float* __restrict__ xz, const float* __restrict__ cw,
                              const float* __restrict__ cb, float* __restrict__ u){
  int i = blockIdx.x*256 + threadIdx.x;                    // B*L*DI = 524288
  if (i >= BB*LL*DI) return;
  int d = i & 511, l = (i>>9)&127, b = i>>16;
  float s = cb[d];
  #pragma unroll
  for (int k=0;k<4;k++){
    int ls = l + k - 3;
    if (ls >= 0) s += cw[d*4+k] * xz[(size_t)(b*LL+ls)*1024 + d];
  }
  u[i] = s * sigf(s);
}

// selective scan: one thread per (b,d), 16 states in registers, fused y + silu gate
__global__ void k_scan(const float* __restrict__ dt, const float* __restrict__ u,
                       const float* __restrict__ dbl, const float* __restrict__ xz,
                       const float* __restrict__ Alog, const float* __restrict__ Dp,
                       float* __restrict__ y){
  int i = blockIdx.x*256 + threadIdx.x;                    // B*DI = 4096
  if (i >= BB*DI) return;
  int d = i & 511, b = i >> 9;
  float A[NST], st[NST];
  #pragma unroll
  for (int n=0;n<NST;n++){ A[n] = -expf(Alog[d*NST+n]); st[n]=0.f; }
  float Dd = Dp[d];
  for (int l=0;l<LL;l++){
    size_t row = (size_t)(b*LL+l);
    float dtv = dt[row*DI + d];
    float uv  = u [row*DI + d];
    float du  = dtv*uv;
    const float* dr = dbl + row*48;
    float acc = 0.f;
    #pragma unroll
    for (int n=0;n<NST;n++){
      float sn = expf(dtv*A[n])*st[n] + du*dr[16+n];       // dA*h + dB*u
      st[n] = sn;
      acc += sn*dr[32+n];                                  // * C
    }
    float zg = xz[row*1024 + 512 + d];
    y[row*DI + d] = (acc + Dd*uv) * (zg * sigf(zg));
  }
}

// in-place LayerNorm over rows of 256
__global__ void k_ln_rows(float* __restrict__ d, const float* __restrict__ g,
                          const float* __restrict__ be){
  int row = blockIdx.x, c = threadIdx.x;
  float v = d[(size_t)row*CC + c];
  __shared__ float s1[256], s2[256];
  s1[c]=v; s2[c]=v*v; __syncthreads();
  for (int o=128;o>0;o>>=1){ if (c<o){s1[c]+=s1[c+o]; s2[c]+=s2[c+o];} __syncthreads(); }
  float m  = s1[0]*(1.f/256.f);
  float vr = s2[0]*(1.f/256.f)-m*m;
  float rs = rsqrtf(vr+1e-5f);
  d[(size_t)row*CC + c] = (v-m)*rs*g[c]+be[c];
}

// cat[b,l,0:256]=fwd[b,l,:]; cat[b,l,256:512]=bwd[b,L-1-l,:]
__global__ void k_concat_flip(const float* __restrict__ f, const float* __restrict__ bk,
                              float* __restrict__ cat){
  int i = blockIdx.x*256 + threadIdx.x;                    // B*L*512
  if (i >= BB*LL*512) return;
  int c = i & 511, l = (i>>9)&127, b = i>>16;
  cat[i] = (c < 256) ? f [((size_t)b*LL+l)*CC + c]
                     : bk[((size_t)b*LL + (LL-1-l))*CC + (c-256)];
}

// res[b,o,i,j] = hardsigmoid(res_max[b,i,o]+res_min[b,j,o]) * z[b,o,i,j]
__global__ void k_res_combine(const float* __restrict__ rm, const float* __restrict__ rn,
                              const float* __restrict__ z, float* __restrict__ res){
  size_t i = (size_t)blockIdx.x*256 + threadIdx.x;
  if (i >= (size_t)BB*CC*H2*W2) return;
  int jj = (int)(i & 127), ii = (int)((i>>7)&127);
  int o  = (int)((i>>14)&255), b = (int)(i>>22);
  float v = rm[((size_t)b*LL+ii)*CC+o] + rn[((size_t)b*LL+jj)*CC+o];
  v = v + 3.f; v = v<0.f?0.f:(v>6.f?6.f:v); v *= (1.f/6.f);
  res[i] = v * z[i];
}

// GroupNorm stats: one block per (b,g); 8 contiguous channels per group
__global__ void k_gn_stats(const float* __restrict__ d, int pix,
                           float* __restrict__ gm, float* __restrict__ gr){
  int bg = blockIdx.x, t = threadIdx.x;
  const float* p = d + (size_t)bg * 8 * pix;
  size_t n = (size_t)8*pix;
  __shared__ float s1[256], s2[256];
  float su=0.f, sq=0.f;
  for (size_t i=t;i<n;i+=256){ float v=p[i]; su+=v; sq+=v*v; }
  s1[t]=su; s2[t]=sq; __syncthreads();
  for (int o=128;o>0;o>>=1){ if (t<o){s1[t]+=s1[t+o]; s2[t]+=s2[t+o];} __syncthreads(); }
  if (t==0){
    float fn = (float)n;
    float m  = s1[0]/fn;
    float vr = s2[0]/fn - m*m;
    gm[bg]=m; gr[bg]=rsqrtf(vr+1e-5f);
  }
}

// downsample ::2,::2 with fused GN apply
__global__ void k_down_gn(const float* __restrict__ res, const float* __restrict__ gm,
                          const float* __restrict__ gr, const float* __restrict__ g,
                          const float* __restrict__ be, float* __restrict__ pool){
  int i = blockIdx.x*256 + threadIdx.x;                    // 8*256*64*64
  if (i >= BB*CC*HH*WW) return;
  int x4 = i & 63, y = (i>>6)&63, c = (i>>12)&255, b = i>>20;
  size_t src = (((size_t)b*CC + c)*H2 + 2*y)*W2 + 2*x4;
  int bg = b*32 + (c>>3);
  pool[i] = (res[src]-gm[bg])*gr[bg]*g[c]+be[c];
}

// GN apply; ADDX: add residual input; OUTSEP: write to separate output
template<bool ADDX, bool OUTSEP>
__global__ void k_gn_apply(float* __restrict__ d, const float* __restrict__ gm,
                           const float* __restrict__ gr, const float* __restrict__ g,
                           const float* __restrict__ be, const float* __restrict__ addx,
                           float* __restrict__ outp, int pix){
  size_t i = (size_t)blockIdx.x*256 + threadIdx.x;
  size_t total = (size_t)BB*CC*pix;
  if (i >= total) return;
  int c = (int)((i/(size_t)pix) % CC);
  int b = (int)(i/((size_t)pix*CC));
  int bg = b*32 + (c>>3);
  float v = (d[i]-gm[bg])*gr[bg]*g[c]+be[c];
  if (ADDX) v += addx[i];
  if (OUTSEP) outp[i]=v; else d[i]=v;
}

// ---------------------------------------------------------------------------
extern "C" void kernel_launch(void* const* d_in, const int* in_sizes, int n_in,
                              void* d_out, int out_size, void* d_ws, size_t ws_size,
                              hipStream_t stream) {
  const float* x        = (const float*)d_in[0];
  const float* pos      = (const float*)d_in[1];
  const float* ln_w     = (const float*)d_in[2];
  const float* ln_b     = (const float*)d_in[3];
  const float* proj_w   = (const float*)d_in[4];
  const float* proj_b   = (const float*)d_in[5];
  const float* up_dw_w  = (const float*)d_in[6];
  const float* up_bn1_g = (const float*)d_in[7];
  const float* up_bn1_b = (const float*)d_in[8];
  const float* up_pw_w  = (const float*)d_in[9];
  const float* up_bn2_g = (const float*)d_in[10];
  const float* up_bn2_b = (const float*)d_in[11];
  const float* down_pw_w= (const float*)d_in[12];
  const float* down_bn_g= (const float*)d_in[13];
  const float* down_bn_b= (const float*)d_in[14];
  const float* gn_g     = (const float*)d_in[15];
  const float* gn_b     = (const float*)d_in[16];
  const float* m_in_w [2] = {(const float*)d_in[17], (const float*)d_in[26]};
  const float* m_cw   [2] = {(const float*)d_in[18], (const float*)d_in[27]};
  const float* m_cb   [2] = {(const float*)d_in[19], (const float*)d_in[28]};
  const float* m_xp   [2] = {(const float*)d_in[20], (const float*)d_in[29]};
  const float* m_dtw  [2] = {(const float*)d_in[21], (const float*)d_in[30]};
  const float* m_dtb  [2] = {(const float*)d_in[22], (const float*)d_in[31]};
  const float* m_Al   [2] = {(const float*)d_in[23], (const float*)d_in[32]};
  const float* m_Dp   [2] = {(const float*)d_in[24], (const float*)d_in[33]};
  const float* m_ow   [2] = {(const float*)d_in[25], (const float*)d_in[34]};

  float* ws = (float*)d_ws;
  size_t off = 0;
  auto alloc = [&](size_t n)->float*{ float* p = ws+off; off += (n+15)&~(size_t)15; return p; };
  float* z0   = alloc(33554432);   // dwconvT out, later reused for res
  float* z    = alloc(33554432);   // up path result
  float* pool = alloc(8388608);
  float* dbuf = alloc(8388608);
  float* xz   = alloc(1048576);
  float* u    = alloc(524288);
  float* dbl  = alloc(49152);
  float* dtb  = alloc(524288);
  float* yb   = alloc(524288);
  float* rmax = alloc(262144);
  float* rmin = alloc(262144);
  float* rmaxf= alloc(262144);
  float* rminf= alloc(262144);
  float* mo0  = alloc(262144);     // fmax
  float* mo1  = alloc(262144);     // fmin
  float* mo2  = alloc(262144);     // bmax
  float* mo3  = alloc(262144);     // bmin
  float* cat  = alloc(524288);
  float* resM = alloc(262144);
  float* resN = alloc(262144);
  float* a1 = alloc(256); float* b1 = alloc(256);
  float* a2 = alloc(256); float* b2 = alloc(256);
  float* a3 = alloc(256); float* b3 = alloc(256);
  float* gm = alloc(256); float* gr = alloc(256);
  bf16* bfp = (bf16*)(ws + off);
  size_t ob = 0;
  auto balloc = [&](size_t n)->bf16*{ bf16* p = bfp+ob; ob += (n+15)&~(size_t)15; return p; };
  bf16* wb_up   = balloc(65536);
  bf16* wb_down = balloc(65536);
  bf16* wb_proj = balloc(131072);
  bf16* wb_in0  = balloc(262144);
  bf16* wb_in1  = balloc(262144);
  bf16* wb_xp0  = balloc(24576);
  bf16* wb_xp1  = balloc(24576);
  bf16* wb_dt0  = balloc(8192);
  bf16* wb_dt1  = balloc(8192);
  bf16* wb_out0 = balloc(131072);
  bf16* wb_out1 = balloc(131072);
  bf16* wb_in [2] = {wb_in0,  wb_in1};
  bf16* wb_xp [2] = {wb_xp0,  wb_xp1};
  bf16* wb_dt [2] = {wb_dt0,  wb_dt1};
  bf16* wb_out[2] = {wb_out0, wb_out1};

  // weight conversions to bf16
  k_cvt_bf16<<<256,256,0,stream>>>(up_pw_w,  wb_up,   65536);
  k_cvt_bf16<<<256,256,0,stream>>>(down_pw_w,wb_down, 65536);
  k_cvt_bf16<<<512,256,0,stream>>>(proj_w,   wb_proj, 131072);
  for (int dir=0; dir<2; ++dir){
    k_cvt_bf16<<<1024,256,0,stream>>>(m_in_w[dir], wb_in[dir],  262144);
    k_cvt_bf16<<<96,  256,0,stream>>>(m_xp[dir],   wb_xp[dir],  24576);
    k_cvt_bf16<<<32,  256,0,stream>>>(m_dtw[dir],  wb_dt[dir],  8192);
    k_cvt_bf16<<<512, 256,0,stream>>>(m_ow[dir],   wb_out[dir], 131072);
  }

  // --- up path: dwconvT -> BN1 stats -> (BN1+ReLU fused) WMMA pw GEMM -> BN2 clip6
  k_dwconvT<<<131072,256,0,stream>>>(x, up_dw_w, z0);
  k_chan_stats<<<256,256,0,stream>>>(z0, 16384, up_bn1_g, up_bn1_b, a1, b1);
  k_gemm_nchw<1><<<dim3(8,512),256,0,stream>>>(z0, wb_up, z, a1, b1, 16384);
  k_chan_stats<<<256,256,0,stream>>>(z, 16384, up_bn2_g, up_bn2_b, a2, b2);
  k_affine<2><<<131072,256,0,stream>>>(z, a2, b2, 16384);

  // --- axial reductions + pos embed + LN
  k_reduce_axial<<<1024,256,0,stream>>>(x, pos, ln_w, ln_b, rmax, rmin);
  k_flip_l<<<1024,256,0,stream>>>(rmax, rmaxf);
  k_flip_l<<<1024,256,0,stream>>>(rmin, rminf);

  // --- 4 mamba invocations: (rmax,f)->mo0 (rmin,f)->mo1 (rmaxf,b)->mo2 (rminf,b)->mo3
  const float* srcs[4] = {rmax, rmin, rmaxf, rminf};
  float* dsts[4] = {mo0, mo1, mo2, mo3};
  for (int iv=0; iv<4; ++iv){
    int dir = iv >> 1;
    k_gemm_seq<0><<<512,256,0,stream>>>(srcs[iv], wb_in[dir], xz, 1024,1024,256,256, nullptr);
    k_conv1d_silu<<<2048,256,0,stream>>>(xz, m_cw[dir], m_cb[dir], u);
    k_gemm_seq<0><<<24, 256,0,stream>>>(u,   wb_xp[dir],  dbl, 1024,48, 512,512, nullptr);
    k_gemm_seq<1><<<256,256,0,stream>>>(dbl, wb_dt[dir],  dtb, 1024,512,16, 48,  m_dtb[dir]);
    k_scan<<<16,256,0,stream>>>(dtb, u, dbl, xz, m_Al[dir], m_Dp[dir], yb);
    k_gemm_seq<0><<<128,256,0,stream>>>(yb,  wb_out[dir], dsts[iv], 1024,256,512,512, nullptr);
    k_ln_rows<<<1024,256,0,stream>>>(dsts[iv], ln_w, ln_b);
  }

  // --- merge fwd/bwd, proj (+bias), LN
  k_concat_flip<<<2048,256,0,stream>>>(mo0, mo2, cat);
  k_gemm_seq<2><<<128,256,0,stream>>>(cat, wb_proj, resM, 1024,256,512,512, proj_b);
  k_ln_rows<<<1024,256,0,stream>>>(resM, ln_w, ln_b);
  k_concat_flip<<<2048,256,0,stream>>>(mo1, mo3, cat);
  k_gemm_seq<2><<<128,256,0,stream>>>(cat, wb_proj, resN, 1024,256,512,512, proj_b);
  k_ln_rows<<<1024,256,0,stream>>>(resN, ln_w, ln_b);

  // --- res = hsig(outer sum) * z; GN stats; fused downsample+GN apply
  k_res_combine<<<131072,256,0,stream>>>(resM, resN, z, z0);   // z0 reused as res
  k_gn_stats<<<256,256,0,stream>>>(z0, 16384, gm, gr);
  k_down_gn<<<32768,256,0,stream>>>(z0, gm, gr, gn_g, gn_b, pool);

  // --- down pw GEMM (WMMA), BN, GN, +x, GN -> d_out
  k_gemm_nchw<0><<<dim3(8,128),256,0,stream>>>(pool, wb_down, dbuf, nullptr,nullptr, 4096);
  k_chan_stats<<<256,256,0,stream>>>(dbuf, 4096, down_bn_g, down_bn_b, a3, b3);
  k_affine<0><<<32768,256,0,stream>>>(dbuf, a3, b3, 4096);
  k_gn_stats<<<256,256,0,stream>>>(dbuf, 4096, gm, gr);
  k_gn_apply<true,false><<<32768,256,0,stream>>>(dbuf, gm, gr, gn_g, gn_b, x, nullptr, 4096);
  k_gn_stats<<<256,256,0,stream>>>(dbuf, 4096, gm, gr);
  k_gn_apply<false,true><<<32768,256,0,stream>>>(dbuf, gm, gr, gn_g, gn_b, nullptr, (float*)d_out, 4096);

  (void)in_sizes; (void)n_in; (void)out_size; (void)ws_size;
}
